// VectorQuantizerEMA_38259568673348
// MI455X (gfx1250) — compile-verified, matrix operand order checked by reading, not run on hardware
//
#include <hip/hip_runtime.h>
#include <hip/hip_bf16.h>
#include <math.h>

// ---------------------------------------------------------------------------
// VQ-VAE EMA vector quantizer for MI455X (gfx1250, wave32, WMMA).
// Sizes fixed by the reference: B=16, D=64, H=64, W=64, K=1024, N=B*H*W=65536.
// ---------------------------------------------------------------------------

#define DECAY 0.99f
#define VQ_EPS 1e-05f

#define BQ 16
#define DQ 64
#define HQ 64
#define WQ 64
#define KQ 1024
#define NQ (BQ * HQ * WQ) // 65536

typedef __attribute__((ext_vector_type(16))) __bf16 v16bf;
typedef __attribute__((ext_vector_type(8)))  float  v8f;

// ---------------------------------------------------------------------------
// Workspace layout (float offsets):
//   ws[0] : loss accumulator          ws[1] : n_total
//   ws[2] : entropy accumulator       ws[3] : pad
//   counts[K], embed_sum[K*D], idx[N] (int), half_enorm[K],
//   pad to 32B, bfmt: WMMA B fragments (64 tiles * 2 chunks * 32 lanes * v16bf)
// ---------------------------------------------------------------------------
#define WS_COUNTS 4
#define WS_ESUM   (WS_COUNTS + KQ)                  // 1028
#define WS_IDX    (WS_ESUM + KQ * DQ)               // 66564
#define WS_ENORM  (WS_IDX + NQ)                     // 132100
#define WS_BFMT   (WS_ENORM + KQ + 4)               // 133128 (32B aligned)
#define WS_ZERO_FLOATS WS_IDX                       // must be zeroed each launch

// Output layout (floats, concatenated in return order):
#define OUT_ZQ    0
#define OUT_LOSS  (OUT_ZQ + BQ * DQ * HQ * WQ)      // 4194304
#define OUT_IDX   (OUT_LOSS + 1)                    // 4194305
#define OUT_PERP  (OUT_IDX + NQ)                    // 4259841
#define OUT_USED  (OUT_PERP + 1)                    // 4259842
#define OUT_NEMB  (OUT_USED + KQ)                   // 4260866
#define OUT_NCS   (OUT_NEMB + KQ * DQ)              // 4326402
#define OUT_NEA   (OUT_NCS + KQ)                    // 4327426

// WMMA 16-bit fragment K index for element e of a v16bf, given lane half.
__device__ __forceinline__ int frag_k(int e, int half) {
  int v = e >> 1;
  return ((v < 4) ? (2 * v) : (16 + 2 * (v - 4))) + 8 * half + (e & 1);
}

// ---------------------------------------------------------------------------
__global__ __launch_bounds__(256) void vq_init(float* ws) {
  int t = blockIdx.x * blockDim.x + threadIdx.x;
  if (t < WS_ZERO_FLOATS) ws[t] = 0.0f;
}

// ---------------------------------------------------------------------------
// Prep: reformat codebook into WMMA B-fragment layout (bf16) once, and
// precompute 0.5*||e_k||^2. One wave per 16-wide k-tile (64 waves total).
// ---------------------------------------------------------------------------
__global__ __launch_bounds__(256) void vq_prep(const float* __restrict__ emb,
                                               v16bf* __restrict__ bfmt,
                                               float* __restrict__ half_enorm) {
  const int lane = threadIdx.x & 31;
  const int wave = threadIdx.x >> 5;
  const int kt   = blockIdx.x * 8 + wave;   // 0..63
  const int m    = lane & 15;
  const int half = lane >> 4;

  const float* erow = emb + (kt * 16 + m) * DQ;
  v16bf b0, b1;
  float en_part = 0.0f;
#pragma unroll
  for (int e = 0; e < 16; ++e) {
    int kk = frag_k(e, half);
    float f0 = erow[kk];
    float f1 = erow[32 + kk];
    b0[e] = (__bf16)f0;
    b1[e] = (__bf16)f1;
    en_part += f0 * f0 + f1 * f1;
  }
  float enorm = en_part + __shfl_xor(en_part, 16, 32);
  bfmt[(kt * 2 + 0) * 32 + lane] = b0;
  bfmt[(kt * 2 + 1) * 32 + lane] = b1;
  if (lane < 16) half_enorm[kt * 16 + m] = 0.5f * enorm;
}

// ---------------------------------------------------------------------------
// Argmin: each wave owns 32 rows (two 16-row A tiles); sweeps the 1024 codes
// in 16-wide tiles from the pre-formatted bf16 fragment buffer.
// argmin(||z||^2+||e||^2-2 z.e) == argmax(z.e - ||e||^2/2), so the hot loop
// is: 4 b128 loads + 1 scalar load + 4 WMMA + 16x(sub,cmp,2 cndmask).
// 256 threads = 8 waves; 256 rows per block; grid = N/256 = 256 blocks.
// ---------------------------------------------------------------------------
__global__ __launch_bounds__(256) void vq_argmin(const float* __restrict__ z,
                                                 const v16bf* __restrict__ bfmt,
                                                 const float* __restrict__ half_enorm,
                                                 int* __restrict__ idx_out,
                                                 float* __restrict__ loss_acc) {
  const int lane = threadIdx.x & 31;
  const int wave = threadIdx.x >> 5;
  const int n0   = (blockIdx.x * 8 + wave) * 32;
  const int m    = lane & 15;
  const int half = lane >> 4;

  // ---- Load two 16-row A tiles (each as two 16x32 bf16 chunks over D=64).
  v16bf A[2][2];
  float zn_c[2][8];
#pragma unroll
  for (int r = 0; r < 2; ++r) {
    const int n = n0 + r * 16 + m;
    const int w = n & 63;
    const int h = (n >> 6) & 63;
    const int b = n >> 12;
    const float* zrow = z + (((b * DQ) * HQ + h) * WQ + w); // + d*(H*W)
    float zn_part = 0.0f;
#pragma unroll
    for (int e = 0; e < 16; ++e) {
      int kk = frag_k(e, half);
      float f0 = zrow[kk * (HQ * WQ)];
      float f1 = zrow[(32 + kk) * (HQ * WQ)];
      A[r][0][e] = (__bf16)f0;
      A[r][1][e] = (__bf16)f1;
      zn_part += f0 * f0 + f1 * f1;
    }
    float znorm = zn_part + __shfl_xor(zn_part, 16, 32);
    // Redistribute to C layout: VGPR i at this lane is row (i + 8*half).
#pragma unroll
    for (int i = 0; i < 8; ++i)
      zn_c[r][i] = __shfl(znorm, i + 8 * half, 32);
  }

  float maxv[2][8];
  int   mink[2][8];
#pragma unroll
  for (int r = 0; r < 2; ++r)
#pragma unroll
    for (int i = 0; i < 8; ++i) { maxv[r][i] = -3.4e38f; mink[r][i] = 0; }

  // ---- Sweep the codebook in 16-wide tiles.
  for (int kt = 0; kt < 64; ++kt) {
    const v16bf* bt = bfmt + kt * 64;
    __builtin_prefetch((const void*)(bt + 64), 0, 3);  // next tile -> global_prefetch_b8
    v16bf b0 = bt[lane];
    v16bf b1 = bt[32 + lane];
    const float he  = half_enorm[kt * 16 + m];  // 0.5*||e_col||^2, col = m
    const int   col = kt * 16 + m;
#pragma unroll
    for (int r = 0; r < 2; ++r) {
      v8f acc = {};
      acc = __builtin_amdgcn_wmma_f32_16x16x32_bf16(false, A[r][0], false, b0,
                                                    (short)0, acc, false, false);
      acc = __builtin_amdgcn_wmma_f32_16x16x32_bf16(false, A[r][1], false, b1,
                                                    (short)0, acc, false, false);
#pragma unroll
      for (int i = 0; i < 8; ++i) {
        float score = acc[i] - he;              // z.e - ||e||^2/2
        if (score > maxv[r][i]) { maxv[r][i] = score; mink[r][i] = col; }
      }
    }
  }

  // ---- Reduce argmax across the 16 columns held by this half-wave.
  float rowloss = 0.0f;
#pragma unroll
  for (int r = 0; r < 2; ++r) {
#pragma unroll
    for (int i = 0; i < 8; ++i) {
      float v = maxv[r][i];
      int   kbest = mink[r][i];
#pragma unroll
      for (int off = 8; off >= 1; off >>= 1) {
        float ov = __shfl_xor(v, off, 32);
        int   ok = __shfl_xor(kbest, off, 32);
        if (ov > v || (ov == v && ok < kbest)) { v = ov; kbest = ok; }
      }
      if (m == 0) {
        idx_out[n0 + r * 16 + i + 8 * half] = kbest;
        rowloss += zn_c[r][i] - 2.0f * v;       // = min ||z - e||^2
      }
    }
  }
  if (m == 0) atomicAdd(loss_acc, rowloss);
}

// ---------------------------------------------------------------------------
// Scatter: z_q output, segment sums (counts, embed_sum), index output.
// One thread per (n, d): N*D = 4.19M threads.
// ---------------------------------------------------------------------------
__global__ __launch_bounds__(256) void vq_scatter(const float* __restrict__ z,
                                                  const float* __restrict__ emb,
                                                  const int* __restrict__ idx,
                                                  float* __restrict__ counts,
                                                  float* __restrict__ embed_sum,
                                                  float* __restrict__ out_zq,
                                                  float* __restrict__ out_idx) {
  int t = blockIdx.x * blockDim.x + threadIdx.x;
  int n = t >> 6;
  int d = t & 63;
  int w = n & 63;
  int h = (n >> 6) & 63;
  int b = n >> 12;
  int zoff = (((b * DQ + d) * HQ + h) * WQ) + w;
  int code = idx[n];
  float zv = z[zoff];
  atomicAdd(&embed_sum[code * DQ + d], zv);
  out_zq[zoff] = emb[code * DQ + d];   // z + sg(q - z) == q numerically
  if (d == 0) {
    atomicAdd(&counts[code], 1.0f);
    out_idx[n] = (float)code;
  }
}

// ---------------------------------------------------------------------------
// EMA pass 1: new_cluster_size, new_embedding_avg, used_codes, n_total,
// entropy accumulation. One thread per (k, d).
// ---------------------------------------------------------------------------
__global__ __launch_bounds__(256) void vq_ema1(const float* __restrict__ counts,
                                               const float* __restrict__ embed_sum,
                                               const float* __restrict__ cluster_size,
                                               const float* __restrict__ embedding_avg,
                                               float* __restrict__ out_ncs,
                                               float* __restrict__ out_nea,
                                               float* __restrict__ out_used,
                                               float* __restrict__ n_total,
                                               float* __restrict__ ent_acc) {
  int t = blockIdx.x * blockDim.x + threadIdx.x;
  int k = t >> 6;
  int d = t & 63;
  float nea = embedding_avg[t] * DECAY + (1.0f - DECAY) * embed_sum[t];
  out_nea[t] = nea;
  if (d == 0) {
    float c   = counts[k];
    float ncs = cluster_size[k] * DECAY + (1.0f - DECAY) * c;
    out_ncs[k]  = ncs;
    out_used[k] = (c > 0.0f) ? 1.0f : 0.0f;
    atomicAdd(n_total, ncs);
    float p = c / (float)NQ;
    atomicAdd(ent_acc, p * logf(p + 1e-10f));
  }
}

// ---------------------------------------------------------------------------
// EMA pass 2: smoothed normalization + scalar outputs.
// ---------------------------------------------------------------------------
__global__ __launch_bounds__(256) void vq_ema2(const float* __restrict__ out_ncs,
                                               const float* __restrict__ out_nea,
                                               float* __restrict__ out_emb,
                                               const float* __restrict__ n_total,
                                               const float* __restrict__ loss_acc,
                                               const float* __restrict__ ent_acc,
                                               float* __restrict__ out_loss,
                                               float* __restrict__ out_perp) {
  int t = blockIdx.x * blockDim.x + threadIdx.x;
  int k = t >> 6;
  float ntot = *n_total;
  float ncs  = out_ncs[k];
  float smoothed = (ncs + VQ_EPS) / (ntot + (float)KQ * VQ_EPS) * ntot;
  out_emb[t] = out_nea[t] / smoothed;
  if (t == 0) {
    *out_loss = *loss_acc / (float)(NQ * DQ);
    *out_perp = expf(-(*ent_acc));
  }
}

// ---------------------------------------------------------------------------
extern "C" void kernel_launch(void* const* d_in, const int* in_sizes, int n_in,
                              void* d_out, int out_size, void* d_ws, size_t ws_size,
                              hipStream_t stream) {
  (void)in_sizes; (void)n_in; (void)out_size; (void)ws_size;

  const float* z             = (const float*)d_in[0];
  const float* embedding     = (const float*)d_in[1];
  const float* cluster_size  = (const float*)d_in[2];
  const float* embedding_avg = (const float*)d_in[3];

  float* out = (float*)d_out;
  float* ws  = (float*)d_ws;

  float* loss_acc   = ws + 0;
  float* n_total    = ws + 1;
  float* ent_acc    = ws + 2;
  float* counts     = ws + WS_COUNTS;
  float* embed_sum  = ws + WS_ESUM;
  int*   idx        = (int*)(ws + WS_IDX);
  float* half_enorm = ws + WS_ENORM;
  v16bf* bfmt       = (v16bf*)(ws + WS_BFMT);

  // 0) zero accumulators / segment sums
  vq_init<<<(WS_ZERO_FLOATS + 255) / 256, 256, 0, stream>>>(ws);

  // 1) one-time codebook reformat to WMMA B fragments + 0.5*||e||^2
  vq_prep<<<8, 256, 0, stream>>>(embedding, bfmt, half_enorm);

  // 2) bf16 WMMA distance GEMM + fused argmin (256 blocks x 8 waves x 32 rows)
  vq_argmin<<<NQ / 256, 256, 0, stream>>>(z, bfmt, half_enorm, idx, loss_acc);

  // 3) z_q, indices, segment sums
  vq_scatter<<<(NQ * DQ) / 256, 256, 0, stream>>>(
      z, embedding, idx, counts, embed_sum, out + OUT_ZQ, out + OUT_IDX);

  // 4) EMA stage 1
  vq_ema1<<<(KQ * DQ) / 256, 256, 0, stream>>>(
      counts, embed_sum, cluster_size, embedding_avg,
      out + OUT_NCS, out + OUT_NEA, out + OUT_USED, n_total, ent_acc);

  // 5) EMA stage 2 + scalars
  vq_ema2<<<(KQ * DQ) / 256, 256, 0, stream>>>(
      out + OUT_NCS, out + OUT_NEA, out + OUT_NEMB,
      n_total, loss_acc, ent_acc, out + OUT_LOSS, out + OUT_PERP);
}
